// ModuleDistLayers_88794153877520
// MI455X (gfx1250) — matrix operand order repaired and checked by reading, not compiled, not run on hardware
//
#include <hip/hip_runtime.h>

#define DEV __device__ __forceinline__

typedef __attribute__((ext_vector_type(16))) __bf16 v16bf;
typedef __attribute__((ext_vector_type(8)))  float  v8f;

constexpr int kN     = 500000;
constexpr int kD     = 128;      // 2*N_AE
constexpr int kNGr   = 4096;
constexpr int kNEle  = 16384;
constexpr int kTiles = kN / 16;  // 31250 (exact)
constexpr float kEps = 1e-5f;

// ---------------- workspace layout (bytes) ----------------
constexpr size_t OFF_ASUM = 0;
constexpr size_t OFF_ACNT = OFF_ASUM + (size_t)kNGr * 64 * 4;
constexpr size_t OFF_ESUM = OFF_ACNT + (size_t)kNGr * 4;
constexpr size_t OFF_ECNT = OFF_ESUM + (size_t)kNEle * 64 * 4;
constexpr size_t OFF_STAT = OFF_ECNT + (size_t)kNEle * 4;
constexpr size_t OFF_ZEND = OFF_STAT + 512 * 4;          // zeroed region end
constexpr size_t OFF_BNP  = OFF_ZEND;
constexpr size_t OFF_PAB  = OFF_BNP + 512 * 4;
constexpr size_t OFF_PEB  = OFF_PAB + (size_t)kNGr * 64 * 2;
constexpr size_t OFF_W1B  = OFF_PEB + (size_t)kNEle * 64 * 2;
constexpr size_t OFF_W2B  = OFF_W1B + (size_t)192 * kD * 2;
constexpr size_t OFF_WFB  = OFF_W2B + (size_t)192 * kD * 2;

DEV unsigned short f2bf(float f) {
  unsigned u = __float_as_uint(f);
  u += 0x7FFFu + ((u >> 16) & 1u);   // round-to-nearest-even
  return (unsigned short)(u >> 16);
}

union ABReg { uint4 q[2]; v16bf v; };

// A-tile: 16x32 bf16, lanes 0-15 -> M=lane, K blocks {0..7,16..23};
// lanes 16-31 -> M=lane-16, K blocks {8..15,24..31}.  (ISA 7.12.2)
DEV v16bf lds_loadA(const unsigned short* tile, int lane, int hb) {
  const int m  = lane & 15;
  const int b0 = (lane >> 4) << 3;
  const unsigned short* p = tile + m * 256 + hb + b0;
  ABReg r;
  r.q[0] = *(const uint4*)(p);
  r.q[1] = *(const uint4*)(p + 16);
  return r.v;
}

// B-tile: 32x16 bf16, lane holds column n = t*16 + lane%16, K blocks {8h..8h+7, 16+8h..16+8h+7}.
DEV v16bf lds_loadB(const unsigned short* W, int lane, int kbase, int t) {
  const unsigned short* p = W + (size_t)(kbase + ((lane >> 4) << 3)) * kD + t * 16 + (lane & 15);
  unsigned u[8];
#pragma unroll
  for (int j = 0; j < 4; ++j) {
    u[j]     = (unsigned)p[(2 * j) * kD] | ((unsigned)p[(2 * j + 1) * kD] << 16);
    u[4 + j] = (unsigned)p[(16 + 2 * j) * kD] | ((unsigned)p[(17 + 2 * j) * kD] << 16);
  }
  ABReg r;
  r.q[0] = make_uint4(u[0], u[1], u[2], u[3]);
  r.q[1] = make_uint4(u[4], u[5], u[6], u[7]);
  return r.v;
}

DEV v8f wmma_bf16(v16bf a, v16bf b, v8f c) {
  return __builtin_amdgcn_wmma_f32_16x16x32_bf16(false, a, false, b, (short)0, c,
                                                 false, false);
}

// Build per-wave h tile [16 rows x 256 bf16]: [rdf(64)|bdf(64)|pa(64)|pe(64)]
DEV void build_htile(unsigned short* h, int lane, int r0,
                     const float* __restrict__ rdf, const float* __restrict__ bdf,
                     const int* __restrict__ aidx, const int* __restrict__ eidx,
                     const unsigned short* __restrict__ paB,
                     const unsigned short* __restrict__ peB) {
  for (int i = lane; i < 256; i += 32) {
    int row = i >> 4;
    int c   = (i & 15) << 2;  // float index 0..60
    float4 a = *(const float4*)(rdf + (size_t)(r0 + row) * 64 + c);
    float4 b = *(const float4*)(bdf + (size_t)(r0 + row) * 64 + c);
    unsigned long long pa =
        (unsigned long long)f2bf(a.x) | ((unsigned long long)f2bf(a.y) << 16) |
        ((unsigned long long)f2bf(a.z) << 32) | ((unsigned long long)f2bf(a.w) << 48);
    unsigned long long pb =
        (unsigned long long)f2bf(b.x) | ((unsigned long long)f2bf(b.y) << 16) |
        ((unsigned long long)f2bf(b.z) << 32) | ((unsigned long long)f2bf(b.w) << 48);
    *(unsigned long long*)(h + row * 256 + c)      = pa;
    *(unsigned long long*)(h + row * 256 + 64 + c) = pb;
  }
  for (int i = lane; i < 128; i += 32) {
    int row = i >> 3;
    int c   = (i & 7) << 3;   // short index, 16B chunks
    int a = aidx[r0 + row];
    int e = eidx[r0 + row];
    *(uint4*)(h + row * 256 + 128 + c) = *(const uint4*)(paB + (size_t)a * 64 + c);
    *(uint4*)(h + row * 256 + 192 + c) = *(const uint4*)(peB + (size_t)e * 64 + c);
  }
}

// y = h @ W.T for one 16x16 output tile t, K=192 (6 WMMA steps).
// h0 selects the dist-feature half (0 = rdf cols, 64 = bdf cols); pooled halves shared.
DEV v8f gemm192(const unsigned short* h, const unsigned short* W, int lane, int t,
                int h0) {
  v8f acc = {0.f, 0.f, 0.f, 0.f, 0.f, 0.f, 0.f, 0.f};
#pragma unroll
  for (int s = 0; s < 6; ++s) {
    int hb = (s < 2) ? (h0 + s * 32) : (128 + (s - 2) * 32);
    v16bf a = lds_loadA(h, lane, hb);
    v16bf b = lds_loadB(W, lane, s * 32, t);
    acc = wmma_bf16(a, b, acc);
  }
  return acc;
}

// ---------------- K1: segment sums (one wave per row) ----------------
__global__ void __launch_bounds__(256)
k_segsum(const float* __restrict__ x, const int* __restrict__ aidx,
         const int* __restrict__ eidx, float* asum, float* acnt, float* esum,
         float* ecnt) {
  int wave = (blockIdx.x * blockDim.x + threadIdx.x) >> 5;
  int lane = threadIdx.x & 31;
  int nw = (gridDim.x * blockDim.x) >> 5;
  for (int r = wave; r < kN; r += nw) {
    if (lane < 16) {
      int a = aidx[r];
      float4 v = *(const float4*)(x + (size_t)r * kD + lane * 4);
      float* d = asum + (size_t)a * 64 + lane * 4;
      atomicAdd(d + 0, v.x); atomicAdd(d + 1, v.y);
      atomicAdd(d + 2, v.z); atomicAdd(d + 3, v.w);
      if (lane == 0) atomicAdd(acnt + a, 1.0f);
    } else {
      int e = eidx[r];
      int j = (lane - 16) * 4;
      float4 v = *(const float4*)(x + (size_t)r * kD + 64 + j);
      float* d = esum + (size_t)e * 64 + j;
      atomicAdd(d + 0, v.x); atomicAdd(d + 1, v.y);
      atomicAdd(d + 2, v.z); atomicAdd(d + 3, v.w);
      if (lane == 16) atomicAdd(ecnt + e, 1.0f);
    }
  }
}

// ---------------- K2: relu(mean) tables + bf16 weight transpose ----------------
__global__ void __launch_bounds__(256)
k_finalize(const float* asum, const float* acnt, const float* esum,
           const float* ecnt, unsigned short* paB, unsigned short* peB,
           const float* W1, const float* W2, const float* Wf,
           unsigned short* w1B, unsigned short* w2B, unsigned short* wfB) {
  int tid = blockIdx.x * blockDim.x + threadIdx.x;
  int stride = gridDim.x * blockDim.x;
  for (int i = tid; i < kNGr * 64; i += stride) {
    float c = fmaxf(acnt[i >> 6], 1.0f);
    paB[i] = f2bf(fmaxf(asum[i] / c, 0.0f));
  }
  for (int i = tid; i < kNEle * 64; i += stride) {
    float c = fmaxf(ecnt[i >> 6], 1.0f);
    peB[i] = f2bf(fmaxf(esum[i] / c, 0.0f));
  }
  for (int i = tid; i < 192 * kD; i += stride) {
    int k = i >> 7, n = i & 127;
    w1B[i] = f2bf(W1[n * 192 + k]);
    w2B[i] = f2bf(W2[n * 192 + k]);
  }
  for (int i = tid; i < 256 * kD; i += stride) {
    int k = i >> 7, n = i & 127;
    wfB[i] = f2bf(Wf[n * 256 + k]);
  }
}

// ---------------- K3: BN statistics pass (WMMA, stats only) ----------------
__global__ void __launch_bounds__(256)
k_ystats(const float* __restrict__ rdf, const float* __restrict__ bdf,
         const int* __restrict__ aidx, const int* __restrict__ eidx,
         const unsigned short* __restrict__ paB,
         const unsigned short* __restrict__ peB,
         const unsigned short* __restrict__ wB,  // w1B|w2B contiguous
         const float* __restrict__ b1, const float* __restrict__ b2,
         float* gstat) {
  extern __shared__ char smem[];
  unsigned short* sW1 = (unsigned short*)smem;      // 192*128
  unsigned short* sW2 = sW1 + 192 * kD;             // 192*128
  unsigned short* sH  = sW2 + 192 * kD;             // 8 waves * 16*256
  float* sStat = (float*)(sH + 8 * 16 * 256);       // 512

  {
    const uint4* src = (const uint4*)wB;
    uint4* dst = (uint4*)smem;
    for (int i = threadIdx.x; i < (2 * 192 * kD) / 8; i += blockDim.x) dst[i] = src[i];
  }
  for (int i = threadIdx.x; i < 512; i += blockDim.x) sStat[i] = 0.0f;
  __syncthreads();

  const int waveId = threadIdx.x >> 5;
  const int lane = threadIdx.x & 31;
  unsigned short* h = sH + waveId * (16 * 256);
  const int colBase = lane & 15;

  for (int tile = blockIdx.x * 8 + waveId; tile < kTiles; tile += gridDim.x * 8) {
    int r0 = tile * 16;
    build_htile(h, lane, r0, rdf, bdf, aidx, eidx, paB, peB);
    for (int t = 0; t < 8; ++t) {
      int col = t * 16 + colBase;
      v8f y1 = gemm192(h, sW1, lane, t, 0);
      float bb = b1[col];
      float s = 0.f, q = 0.f;
#pragma unroll
      for (int v = 0; v < 8; ++v) { float y = y1[v] + bb; s += y; q += y * y; }
      atomicAdd(&sStat[col], s);
      atomicAdd(&sStat[kD + col], q);
      v8f y2 = gemm192(h, sW2, lane, t, 64);
      bb = b2[col];
      s = 0.f; q = 0.f;
#pragma unroll
      for (int v = 0; v < 8; ++v) { float y = y2[v] + bb; s += y; q += y * y; }
      atomicAdd(&sStat[2 * kD + col], s);
      atomicAdd(&sStat[3 * kD + col], q);
    }
  }
  __syncthreads();
  for (int i = threadIdx.x; i < 512; i += blockDim.x) atomicAdd(&gstat[i], sStat[i]);
}

// ---------------- K4: fold BN into per-column scale/shift ----------------
__global__ void k_bnfin(const float* stats, const float* g1, const float* be1,
                        const float* g2, const float* be2, float* bnp) {
  int i = threadIdx.x;
  if (i < 256) {
    int layer = i >> 7, col = i & 127;
    float sum = stats[layer * 256 + col];
    float sq  = stats[layer * 256 + kD + col];
    const float invN = 1.0f / (float)kN;
    float mu  = sum * invN;
    float var = fmaxf(sq * invN - mu * mu, 0.0f);
    float g  = layer ? g2[col] : g1[col];
    float be = layer ? be2[col] : be1[col];
    float sc = g * __frsqrt_rn(var + kEps);
    bnp[layer * 256 + col]      = sc;
    bnp[layer * 256 + kD + col] = be - mu * sc;
  }
}

// ---------------- K5: fused recompute + BN + residual + FC ----------------
__global__ void __launch_bounds__(256)
k_final(const float* __restrict__ x, const float* __restrict__ rdf,
        const float* __restrict__ bdf, const int* __restrict__ aidx,
        const int* __restrict__ eidx, const unsigned short* __restrict__ paB,
        const unsigned short* __restrict__ peB,
        const unsigned short* __restrict__ wB,  // w1B|w2B|wfB contiguous
        const float* __restrict__ b1, const float* __restrict__ b2,
        const float* __restrict__ bf, const float* __restrict__ bnp,
        float* __restrict__ out) {
  extern __shared__ char smem[];
  unsigned short* sW1 = (unsigned short*)smem;   // 192*128
  unsigned short* sW2 = sW1 + 192 * kD;          // 192*128
  unsigned short* sWf = sW2 + 192 * kD;          // 256*128
  unsigned short* sH  = sWf + 256 * kD;          // 8*16*256
  unsigned short* sX  = sH + 8 * 16 * 256;       // 8*16*256

  {
    const uint4* src = (const uint4*)wB;
    uint4* dst = (uint4*)smem;
    const int n16 = (2 * 192 * kD + 256 * kD) / 8;
    for (int i = threadIdx.x; i < n16; i += blockDim.x) dst[i] = src[i];
  }
  __syncthreads();

  const int waveId = threadIdx.x >> 5;
  const int lane = threadIdx.x & 31;
  unsigned short* h  = sH + waveId * (16 * 256);
  unsigned short* xt = sX + waveId * (16 * 256);
  const int colBase = lane & 15;
  const int mHi = (lane >> 4) << 3;

  for (int tile = blockIdx.x * 8 + waveId; tile < kTiles; tile += gridDim.x * 8) {
    int r0 = tile * 16;
    build_htile(h, lane, r0, rdf, bdf, aidx, eidx, paB, peB);

    for (int t = 0; t < 8; ++t) {
      int col = t * 16 + colBase;
      float xv[8];
#pragma unroll
      for (int v = 0; v < 8; ++v)
        xv[v] = x[(size_t)(r0 + mHi + v) * kD + col];

      v8f y1 = gemm192(h, sW1, lane, t, 0);
      {
        float bb = b1[col], sc = bnp[col], sh = bnp[kD + col];
#pragma unroll
        for (int v = 0; v < 8; ++v) {
          float r = fmaxf(sc * (y1[v] + bb) + sh + xv[v], 0.0f);
          xt[(mHi + v) * 256 + col] = f2bf(r);
        }
      }
      v8f y2 = gemm192(h, sW2, lane, t, 64);
      {
        float bb = b2[col], sc = bnp[2 * kD + col], sh = bnp[3 * kD + col];
#pragma unroll
        for (int v = 0; v < 8; ++v) {
          float r = fmaxf(sc * (y2[v] + bb) + sh + xv[v], 0.0f);
          xt[(mHi + v) * 256 + 128 + col] = f2bf(r);
        }
      }
    }

    // FC: [16x256] @ [256x128], K = 8 WMMA steps, A reused across 8 N-tiles.
    v8f accF[8];
    const v8f zero = {0.f, 0.f, 0.f, 0.f, 0.f, 0.f, 0.f, 0.f};
#pragma unroll
    for (int t = 0; t < 8; ++t) accF[t] = zero;
#pragma unroll
    for (int s = 0; s < 8; ++s) {
      v16bf a = lds_loadA(xt, lane, s * 32);
#pragma unroll
      for (int t = 0; t < 8; ++t) {
        v16bf b = lds_loadB(sWf, lane, s * 32, t);
        accF[t] = wmma_bf16(a, b, accF[t]);
      }
    }
#pragma unroll
    for (int t = 0; t < 8; ++t) {
      int col = t * 16 + colBase;
      float bb = bf[col];
#pragma unroll
      for (int v = 0; v < 8; ++v)
        out[(size_t)(r0 + mHi + v) * kD + col] = fmaxf(accF[t][v] + bb, 0.0f);
    }
  }
}

extern "C" void kernel_launch(void* const* d_in, const int* in_sizes, int n_in,
                              void* d_out, int out_size, void* d_ws, size_t ws_size,
                              hipStream_t stream) {
  const float* x   = (const float*)d_in[0];
  const float* rdf = (const float*)d_in[1];
  const float* bdf = (const float*)d_in[2];
  const int* aidx  = (const int*)d_in[3];
  const int* eidx  = (const int*)d_in[4];
  const float* W1  = (const float*)d_in[5];
  const float* b1  = (const float*)d_in[6];
  const float* g1  = (const float*)d_in[7];
  const float* be1 = (const float*)d_in[8];
  const float* W2  = (const float*)d_in[9];
  const float* b2  = (const float*)d_in[10];
  const float* g2  = (const float*)d_in[11];
  const float* be2 = (const float*)d_in[12];
  const float* Wf  = (const float*)d_in[13];
  const float* bff = (const float*)d_in[14];
  float* out = (float*)d_out;

  char* ws = (char*)d_ws;
  float* asum = (float*)(ws + OFF_ASUM);
  float* acnt = (float*)(ws + OFF_ACNT);
  float* esum = (float*)(ws + OFF_ESUM);
  float* ecnt = (float*)(ws + OFF_ECNT);
  float* gstat = (float*)(ws + OFF_STAT);
  float* bnp = (float*)(ws + OFF_BNP);
  unsigned short* paB = (unsigned short*)(ws + OFF_PAB);
  unsigned short* peB = (unsigned short*)(ws + OFF_PEB);
  unsigned short* w1B = (unsigned short*)(ws + OFF_W1B);
  unsigned short* w2B = (unsigned short*)(ws + OFF_W2B);
  unsigned short* wfB = (unsigned short*)(ws + OFF_WFB);

  hipMemsetAsync(ws, 0, OFF_ZEND, stream);

  k_segsum<<<2048, 256, 0, stream>>>(x, aidx, eidx, asum, acnt, esum, ecnt);
  k_finalize<<<1024, 256, 0, stream>>>(asum, acnt, esum, ecnt, paB, peB, W1, W2,
                                       Wf, w1B, w2B, wfB);
  const size_t smem3 = (size_t)(2 * 192 * kD + 8 * 16 * 256) * 2 + 512 * 4;
  k_ystats<<<1024, 256, smem3, stream>>>(rdf, bdf, aidx, eidx, paB, peB, w1B, b1,
                                         b2, gstat);
  k_bnfin<<<1, 256, 0, stream>>>(gstat, g1, be1, g2, be2, bnp);
  const size_t smem5 = (size_t)(2 * 192 * kD + 256 * kD + 2 * 8 * 16 * 256) * 2;
  k_final<<<1024, 256, smem5, stream>>>(x, rdf, bdf, aidx, eidx, paB, peB, w1B,
                                        b1, b2, bff, bnp, out);
}
